// GKAN_Nodes_1047972021083
// MI455X (gfx1250) — matrix operand order repaired and checked by reading, not compiled, not run on hardware
//
#include <hip/hip_runtime.h>
#include <hip/hip_bf16.h>
#include <stdint.h>

// ---------------------------------------------------------------------------
// GKAN (GCNConv + KAN head) forward for MI455X / gfx1250.
// - all matmuls on v_wmma_f32_16x16x32_bf16 (f32 accumulate)
// - weight panel parked in LDS once per block via TDM tensor_load_to_lds
//   (TENSORcnt / s_wait_tensorcnt), with TDM row padding to dodge LDS bank
//   conflicts; A tiles double-buffered -> one barrier per K step
// - KAN head fuses silu + 7 cubic B-spline bases as augmented K dim (x8)
// ---------------------------------------------------------------------------

typedef __attribute__((ext_vector_type(16))) __bf16 v16bf;
typedef __attribute__((ext_vector_type(8)))  __bf16 v8bf;
typedef __attribute__((ext_vector_type(8)))  float  v8f;
typedef __attribute__((ext_vector_type(4)))  unsigned int v4u;
typedef __attribute__((ext_vector_type(8)))  int v8i;
typedef __attribute__((ext_vector_type(4)))  int v4i;

namespace {
constexpr int N_NODES = 16000;
constexpr int N_EDGES = 400000;
constexpr int F_IN    = 500;
constexpr int H_DIM   = 256;
constexpr int C_OUT   = 64;
constexpr int DIN     = F_IN + 3 * H_DIM;   // 1268
constexpr int KAUG    = DIN * 8;            // 10144 (silu + 7 spline bases)
}

__device__ __forceinline__ unsigned short f2bfu(float f) {
  unsigned int u = __builtin_bit_cast(unsigned int, f);
  unsigned int r = u + 0x7FFFu + ((u >> 16) & 1u);   // round-to-nearest-even
  return (unsigned short)(r >> 16);
}

__device__ __forceinline__ float gelu_exact(float v) {
  return 0.5f * v * (1.0f + erff(v * 0.70710678118654752f));
}

__device__ __forceinline__ float silu(float v) {
  return v / (1.0f + __expf(-v));
}

// Cox-de Boor order-3 bases on uniform grid g[t] = -2.5 + 0.5*t, t=0..10
__device__ __forceinline__ void spline7(float z, float* out7) {
  float b[10];
#pragma unroll
  for (int t = 0; t < 10; ++t) {
    float g0 = -2.5f + 0.5f * (float)t;
    b[t] = (z >= g0 && z < g0 + 0.5f) ? 1.0f : 0.0f;
  }
#pragma unroll
  for (int j = 1; j <= 3; ++j) {
    float inv = 1.0f / (0.5f * (float)j);
#pragma unroll
    for (int t = 0; t < 9; ++t) {
      if (t < 10 - j) {
        float gt  = -2.5f + 0.5f * (float)t;
        float gtj = gt + 0.5f * (float)j;
        b[t] = (z - gt) * inv * b[t] + (gtj + 0.5f - z) * inv * b[t + 1];
      }
    }
  }
#pragma unroll
  for (int t = 0; t < 7; ++t) out7[t] = b[t];
}

// ---------------------------------------------------------------------------
// small utility kernels
// ---------------------------------------------------------------------------
__global__ __launch_bounds__(256) void k_fill(float* __restrict__ p, float v, int n) {
  int i = blockIdx.x * blockDim.x + threadIdx.x;
  if (i < n) p[i] = v;
}

__global__ __launch_bounds__(256) void k_deg_edges(const int* __restrict__ ei,
                                                   float* __restrict__ deg) {
  int e = blockIdx.x * blockDim.x + threadIdx.x;
  if (e < N_EDGES) atomicAdd(&deg[ei[N_EDGES + e]], 1.0f);
}

__global__ __launch_bounds__(256) void k_rsqrt_inplace(float* __restrict__ p, int n) {
  int i = blockIdx.x * blockDim.x + threadIdx.x;
  if (i < n) p[i] = rsqrtf(p[i]);
}

__global__ __launch_bounds__(256) void k_copy_x(const float* __restrict__ x,
                                                float* __restrict__ hcat) {
  int i = blockIdx.x * blockDim.x + threadIdx.x;
  if (i < N_NODES * F_IN) {
    int n = i / F_IN, f = i - n * F_IN;
    hcat[(size_t)n * DIN + f] = x[i];
  }
}

// cast fp32 [rows x ld] (cols 0..K) -> bf16 [rows x Kpad], zero-padded; mode 1 = gelu
__global__ __launch_bounds__(256) void k_cast_bf16(const float* __restrict__ src, int ld,
                                                   int K, int Kpad,
                                                   unsigned short* __restrict__ dst,
                                                   int mode, int total) {
  int i = blockIdx.x * blockDim.x + threadIdx.x;
  if (i >= total) return;
  int n = i / Kpad, k = i - n * Kpad;
  float v = (k < K) ? src[(size_t)n * ld + k] : 0.0f;
  if (mode == 1) v = gelu_exact(v);
  dst[i] = f2bfu(v);
}

// w [Nout x K] fp32 -> Bt [Npad x Kpad] bf16 (zero-padded)
__global__ __launch_bounds__(256) void k_prep_w(const float* __restrict__ w,
                                                unsigned short* __restrict__ Bt,
                                                int Nout, int K, int Kpad, int total) {
  int i = blockIdx.x * blockDim.x + threadIdx.x;
  if (i >= total) return;
  int n = i / Kpad, k = i - n * Kpad;
  float v = (n < Nout && k < K) ? w[(size_t)n * K + k] : 0.0f;
  Bt[i] = f2bfu(v);
}

// augmented head weights: Bt[n][i*8+0] = base_w[n,i]; Bt[n][i*8+1+g] = spline_w[n,i,g]
__global__ __launch_bounds__(256) void k_prep_head_w(const float* __restrict__ base_w,
                                                     const float* __restrict__ spline_w,
                                                     unsigned short* __restrict__ Bt) {
  int i = blockIdx.x * blockDim.x + threadIdx.x;
  if (i >= C_OUT * KAUG) return;
  int n = i / KAUG, ka = i - n * KAUG;
  int fi = ka >> 3, j = ka & 7;
  float v = (j == 0) ? base_w[(size_t)n * DIN + fi]
                     : spline_w[((size_t)n * DIN + fi) * 7 + (j - 1)];
  Bt[i] = f2bfu(v);
}

// agg[n,c] = cbias[c] + z[n,c] * dinv[n]^2  (self loop fused, single writer)
__global__ __launch_bounds__(256) void k_agg_init(const float* __restrict__ z,
                                                  const float* __restrict__ dinv,
                                                  const float* __restrict__ cbias,
                                                  float* __restrict__ agg) {
  int i = blockIdx.x * blockDim.x + threadIdx.x;
  if (i >= N_NODES * H_DIM) return;
  int n = i >> 8, c = i & 255;
  float di = dinv[n];
  agg[i] = cbias[c] + z[i] * di * di;
}

// edge scatter: one thread per (edge, 8-channel chunk)
__global__ __launch_bounds__(256) void k_edge_scatter(const int* __restrict__ ei,
                                                      const float* __restrict__ z,
                                                      const float* __restrict__ dinv,
                                                      float* __restrict__ agg) {
  long long gid = (long long)blockIdx.x * blockDim.x + threadIdx.x;
  if (gid >= (long long)N_EDGES * 32) return;
  int e  = (int)(gid >> 5);
  int c0 = ((int)gid & 31) * 8;
  int s = ei[e], d = ei[N_EDGES + e];
  float w = dinv[s] * dinv[d];
  const float4* zp = (const float4*)(z + (size_t)s * H_DIM + c0);
  float4 a = zp[0], b = zp[1];
  float* ap = agg + (size_t)d * H_DIM + c0;
  atomicAdd(ap + 0, a.x * w); atomicAdd(ap + 1, a.y * w);
  atomicAdd(ap + 2, a.z * w); atomicAdd(ap + 3, a.w * w);
  atomicAdd(ap + 4, b.x * w); atomicAdd(ap + 5, b.y * w);
  atomicAdd(ap + 6, b.z * w); atomicAdd(ap + 7, b.w * w);
}

// BN partial sums: block handles 64 rows, thread t owns column t (coalesced)
__global__ __launch_bounds__(256) void k_bn_partial(const float* __restrict__ agg,
                                                    float* __restrict__ musum,
                                                    float* __restrict__ qsum) {
  int c = threadIdx.x;
  int r0 = blockIdx.x * 64;
  float s = 0.0f, q = 0.0f;
#pragma unroll 4
  for (int j = 0; j < 64; ++j) {
    float v = agg[(size_t)(r0 + j) * H_DIM + c];
    s += v; q += v * v;
  }
  atomicAdd(&musum[c], s);
  atomicAdd(&qsum[c], q);
}

__global__ __launch_bounds__(256) void k_bn_finalize(float* __restrict__ musum,
                                                     float* __restrict__ qsum) {
  int c = threadIdx.x;
  float mu = musum[c] * (1.0f / (float)N_NODES);
  float var = qsum[c] * (1.0f / (float)N_NODES) - mu * mu;
  musum[c] = mu;
  qsum[c]  = rsqrtf(var + 1e-5f);
}

__global__ __launch_bounds__(256) void k_bn_apply(const float* __restrict__ agg,
                                                  const float* __restrict__ mu,
                                                  const float* __restrict__ rstd,
                                                  const float* __restrict__ g,
                                                  const float* __restrict__ b,
                                                  float* __restrict__ hcat, int coloff) {
  int i = blockIdx.x * blockDim.x + threadIdx.x;
  if (i >= N_NODES * H_DIM) return;
  int n = i >> 8, c = i & 255;
  float v = (agg[i] - mu[c]) * rstd[c] * g[c] + b[c];
  hcat[(size_t)n * DIN + coloff + c] = v;
}

// ---------------------------------------------------------------------------
// WMMA GEMM: C[M x ldc] = A_bf16[M x Kpad] * Bt_bf16[Npad x Kpad]^T + bias
// block = 128 threads (4 waves), 64x64 output tile, 32-wide K steps.
// B panel [64 x Kpad] loaded ONCE per block into LDS by the Tensor Data Mover
// with 16B/row padding (bank-conflict avoidance); A tiles double-buffered.
// dynamic LDS = 64*(Kpad+8)*2 + 2*64*32*2 bytes.
// ---------------------------------------------------------------------------
__global__ __launch_bounds__(128) void k_wmma_gemm(const unsigned short* __restrict__ A,
                                                   const unsigned short* __restrict__ Bt,
                                                   const float* __restrict__ bias,
                                                   float* __restrict__ C,
                                                   int Kpad, int Nout, int ldc) {
  extern __shared__ __align__(16) unsigned short smem[];
  const int ldb = Kpad + 8;                       // +16B TDM pad per row
  unsigned short* Bs = smem;                      // [64][ldb]
  unsigned short* As = smem + 64 * ldb;           // [2][64][32]
  const int tid  = threadIdx.x;
  const int wave = tid >> 5;
  const int lane = tid & 31;
  const int m0 = blockIdx.y * 64;
  const int n0 = blockIdx.x * 64;

  // ---- TDM: DMA the whole B panel into LDS (wave 0 only; TDM ignores EXEC)
  if (tid < 32) {
    unsigned long long gaddr =
        (unsigned long long)(uintptr_t)(const void*)(Bt + (size_t)n0 * Kpad);
    unsigned laddr = (unsigned)(uintptr_t)(void*)Bs;
    v4u g0;
    g0.x = 1u;                                            // count=1, valid
    g0.y = laddr;                                         // lds_addr
    g0.z = (unsigned)(gaddr & 0xFFFFFFFFu);               // global_addr[31:0]
    g0.w = ((unsigned)(gaddr >> 32) & 0x1FFFFFFu) | (2u << 30);  // [56:32] | type=2
    const unsigned rowDw = (unsigned)(Kpad >> 1);         // dwords per row
    const unsigned padIv = (rowDw == 256u) ? 7u : 6u;     // 2^(v+1) dwords
    const unsigned padAm = 3u;                            // 4 dwords = 16B
    v8i g1;
    g1[0] = (int)((1u << 16) | (1u << 20) | (padIv << 22) | (padAm << 25));
    g1[1] = (int)((unsigned)Kpad << 16);                  // tensor_dim0[15:0]
    g1[2] = (int)(64u << 16);                             // tensor_dim1[15:0]
    g1[3] = (int)((unsigned)Kpad << 16);                  // tile_dim0
    g1[4] = (int)64u;                                     // tile_dim1 (tile_dim2=0)
    g1[5] = (int)Kpad;                                    // tensor_dim0_stride[31:0]
    g1[6] = 0;
    g1[7] = 0;
    v4i zz4 = {0, 0, 0, 0};
    v8i zz8 = {0, 0, 0, 0, 0, 0, 0, 0};
    __builtin_amdgcn_tensor_load_to_lds(g0, g1, zz4, zz4, zz8, 0);
    __builtin_amdgcn_s_wait_tensorcnt(0);
  }
  __syncthreads();   // publish B panel to all waves

  v8f acc[4] = {};
  const int nk = Kpad >> 5;
  for (int kt = 0; kt < nk; ++kt) {
    unsigned short* Ab = As + (kt & 1) * (64 * 32);
#pragma unroll
    for (int r = 0; r < 2; ++r) {
      int idx = tid + (r << 7);
      int row = idx >> 2;
      int ch  = idx & 3;
      const unsigned short* ap = A + (size_t)(m0 + row) * Kpad + (kt << 5) + (ch << 3);
      *(uint4*)&Ab[row * 32 + (ch << 3)] = *(const uint4*)ap;
      __builtin_prefetch(ap + 32, 0, 1);    // speculative: next K tile
    }
    __syncthreads();
    const int mrow = (wave << 4) + (lane & 15);
    const int ka0  = (lane >> 4) << 4;
    v8bf alo = *(const v8bf*)&Ab[mrow * 32 + ka0];
    v8bf ahi = *(const v8bf*)&Ab[mrow * 32 + ka0 + 8];
    v16bf afrag = __builtin_shufflevector(alo, ahi,
        0, 1, 2, 3, 4, 5, 6, 7, 8, 9, 10, 11, 12, 13, 14, 15);
    const int kb = (kt << 5) + ((lane >> 4) << 4);
#pragma unroll
    for (int f = 0; f < 4; ++f) {
      const int ncol = (f << 4) + (lane & 15);
      v16bf bfrag = *(const v16bf*)&Bs[ncol * ldb + kb];
      acc[f] = __builtin_amdgcn_wmma_f32_16x16x32_bf16(
          false, afrag, false, bfrag, (short)0, acc[f], false, false);
    }
  }

#pragma unroll
  for (int f = 0; f < 4; ++f) {
    int n = n0 + (f << 4) + (lane & 15);
    if (n < Nout) {
      float bv = bias ? bias[n] : 0.0f;
#pragma unroll
      for (int r = 0; r < 8; ++r) {
        int m = m0 + (wave << 4) + r + ((lane >> 4) << 3);
        C[(size_t)m * ldc + n] = acc[f][r] + bv;
      }
    }
  }
}

// ---------------------------------------------------------------------------
// KAN head GEMM: out[16000 x 64] = AUG(hcat)[16000 x 10144] * Bt[64 x 10144]^T
// augmented A tile built on the fly (silu + 7 spline bases per feature);
// A and B tiles double-buffered -> one barrier per K step.
// ---------------------------------------------------------------------------
__global__ __launch_bounds__(128) void k_wmma_head(const float* __restrict__ hcat,
                                                   const unsigned short* __restrict__ Bt,
                                                   float* __restrict__ out) {
  __shared__ __align__(16) unsigned short As[2][64][32];
  __shared__ __align__(16) unsigned short Bs[2][64][32];
  const int tid  = threadIdx.x;
  const int wave = tid >> 5;
  const int lane = tid & 31;
  const int m0 = blockIdx.y * 64;

  v8f acc[4] = {};
  const int nk = KAUG >> 5;   // 317; each tile covers 4 input features
  for (int kt = 0; kt < nk; ++kt) {
    const int p = kt & 1;
#pragma unroll
    for (int r = 0; r < 2; ++r) {
      int idx = tid + (r << 7);
      int row = idx >> 2;
      int fi  = idx & 3;
      int feat = kt * 4 + fi;
      float z = hcat[(size_t)(m0 + row) * DIN + feat];
      float v[8];
      v[0] = silu(z);
      spline7(z, v + 1);
      uint4 pk;
      pk.x = (unsigned)f2bfu(v[0]) | ((unsigned)f2bfu(v[1]) << 16);
      pk.y = (unsigned)f2bfu(v[2]) | ((unsigned)f2bfu(v[3]) << 16);
      pk.z = (unsigned)f2bfu(v[4]) | ((unsigned)f2bfu(v[5]) << 16);
      pk.w = (unsigned)f2bfu(v[6]) | ((unsigned)f2bfu(v[7]) << 16);
      *(uint4*)&As[p][row][fi << 3] = pk;
      const unsigned short* bp = Bt + (size_t)row * KAUG + (kt << 5) + (fi << 3);
      *(uint4*)&Bs[p][row][fi << 3] = *(const uint4*)bp;
      __builtin_prefetch(bp + 32, 0, 1);
    }
    __syncthreads();
    const int mrow = (wave << 4) + (lane & 15);
    const int ka0  = (lane >> 4) << 4;
    v8bf alo = *(const v8bf*)&As[p][mrow][ka0];
    v8bf ahi = *(const v8bf*)&As[p][mrow][ka0 + 8];
    v16bf afrag = __builtin_shufflevector(alo, ahi,
        0, 1, 2, 3, 4, 5, 6, 7, 8, 9, 10, 11, 12, 13, 14, 15);
    const int kb0 = (lane >> 4) << 4;
#pragma unroll
    for (int f = 0; f < 4; ++f) {
      const int ncol = (f << 4) + (lane & 15);
      v16bf bfrag = *(const v16bf*)&Bs[p][ncol][kb0];
      acc[f] = __builtin_amdgcn_wmma_f32_16x16x32_bf16(
          false, afrag, false, bfrag, (short)0, acc[f], false, false);
    }
  }

#pragma unroll
  for (int f = 0; f < 4; ++f) {
    int n = (f << 4) + (lane & 15);
#pragma unroll
    for (int r = 0; r < 8; ++r) {
      int m = m0 + (wave << 4) + r + ((lane >> 4) << 3);
      out[(size_t)m * C_OUT + n] = acc[f][r];
    }
  }
}

// ---------------------------------------------------------------------------
extern "C" void kernel_launch(void* const* d_in, const int* in_sizes, int n_in,
                              void* d_out, int out_size, void* d_ws, size_t ws_size,
                              hipStream_t stream) {
  (void)in_sizes; (void)n_in; (void)out_size; (void)ws_size;

  const float* x = (const float*)d_in[0];
  const float* fc1w[3] = {(const float*)d_in[1],  (const float*)d_in[8],  (const float*)d_in[15]};
  const float* fc1b[3] = {(const float*)d_in[2],  (const float*)d_in[9],  (const float*)d_in[16]};
  const float* fc2w[3] = {(const float*)d_in[3],  (const float*)d_in[10], (const float*)d_in[17]};
  const float* fc2b[3] = {(const float*)d_in[4],  (const float*)d_in[11], (const float*)d_in[18]};
  const float* cbias[3]= {(const float*)d_in[5],  (const float*)d_in[12], (const float*)d_in[19]};
  const float* bng[3]  = {(const float*)d_in[6],  (const float*)d_in[13], (const float*)d_in[20]};
  const float* bnb[3]  = {(const float*)d_in[7],  (const float*)d_in[14], (const float*)d_in[21]};
  const float* base_w   = (const float*)d_in[22];
  const float* spline_w = (const float*)d_in[23];
  const int*   ei       = (const int*)d_in[24];
  float* out = (float*)d_out;

  // workspace carve-out (256B aligned)
  char* ws = (char*)d_ws;
  size_t off = 0;
  auto carve = [&](size_t bytes) {
    void* p = ws + off;
    off = (off + bytes + 255) & ~(size_t)255;
    return p;
  };
  float*          dinv  = (float*)carve((size_t)N_NODES * 4);
  float*          hcat  = (float*)carve((size_t)N_NODES * DIN * 4);
  float*          z1    = (float*)carve((size_t)N_NODES * 512 * 4);
  float*          z2    = (float*)carve((size_t)N_NODES * H_DIM * 4);
  float*          agg   = (float*)carve((size_t)N_NODES * H_DIM * 4);
  unsigned short* Abuf  = (unsigned short*)carve((size_t)N_NODES * 512 * 2);
  unsigned short* Btb   = (unsigned short*)carve((size_t)C_OUT * KAUG * 2);
  float*          musum = (float*)carve(256 * 4);
  float*          qsum  = (float*)carve(256 * 4);

  // degree -> dinv
  k_fill<<<(N_NODES + 255) / 256, 256, 0, stream>>>(dinv, 1.0f, N_NODES);
  k_deg_edges<<<(N_EDGES + 255) / 256, 256, 0, stream>>>(ei, dinv);
  k_rsqrt_inplace<<<(N_NODES + 255) / 256, 256, 0, stream>>>(dinv, N_NODES);

  // hcat[:, 0:500] = x
  k_copy_x<<<(N_NODES * F_IN + 255) / 256, 256, 0, stream>>>(x, hcat);

  for (int i = 0; i < 3; ++i) {
    const int fin  = (i == 0) ? F_IN : H_DIM;
    const int Kpad = (i == 0) ? 512  : 256;
    const int NpF  = (i == 0) ? 512  : 256;
    const float* src = (i == 0) ? hcat : (hcat + F_IN + (size_t)(i - 1) * H_DIM);
    const size_t shb = (size_t)64 * (Kpad + 8) * 2 + (size_t)2 * 64 * 32 * 2;

    // fc1
    int totA = N_NODES * Kpad;
    k_cast_bf16<<<(totA + 255) / 256, 256, 0, stream>>>(src, DIN, fin, Kpad, Abuf, 0, totA);
    int totW1 = NpF * Kpad;
    k_prep_w<<<(totW1 + 255) / 256, 256, 0, stream>>>(fc1w[i], Btb, fin, fin, Kpad, totW1);
    {
      dim3 g(NpF / 64, N_NODES / 64);
      k_wmma_gemm<<<g, 128, shb, stream>>>(Abuf, Btb, fc1b[i], z1, Kpad, fin, fin);
    }
    // gelu + fc2
    k_cast_bf16<<<(totA + 255) / 256, 256, 0, stream>>>(z1, fin, fin, Kpad, Abuf, 1, totA);
    int totW2 = 256 * Kpad;
    k_prep_w<<<(totW2 + 255) / 256, 256, 0, stream>>>(fc2w[i], Btb, H_DIM, fin, Kpad, totW2);
    {
      dim3 g(256 / 64, N_NODES / 64);
      k_wmma_gemm<<<g, 128, shb, stream>>>(Abuf, Btb, fc2b[i], z2, Kpad, H_DIM, H_DIM);
    }

    // GCN aggregation (self loop + bias fused into init, then edge scatter)
    k_agg_init<<<(N_NODES * H_DIM + 255) / 256, 256, 0, stream>>>(z2, dinv, cbias[i], agg);
    {
      long long tot = (long long)N_EDGES * 32;
      k_edge_scatter<<<(unsigned)((tot + 255) / 256), 256, 0, stream>>>(ei, z2, dinv, agg);
    }

    // BatchNorm (batch stats) -> hcat[:, F + i*H]
    k_fill<<<1, 256, 0, stream>>>(musum, 0.0f, 256);
    k_fill<<<1, 256, 0, stream>>>(qsum, 0.0f, 256);
    k_bn_partial<<<N_NODES / 64, 256, 0, stream>>>(agg, musum, qsum);
    k_bn_finalize<<<1, 256, 0, stream>>>(musum, qsum);
    k_bn_apply<<<(N_NODES * H_DIM + 255) / 256, 256, 0, stream>>>(
        agg, musum, qsum, bng[i], bnb[i], hcat, F_IN + i * H_DIM);
  }

  // KAN head: augmented GEMM 16000 x 10144 x 64
  k_prep_head_w<<<(C_OUT * KAUG + 255) / 256, 256, 0, stream>>>(base_w, spline_w, Btb);
  {
    dim3 g(1, N_NODES / 64);
    k_wmma_head<<<g, 128, 0, stream>>>(hcat, Btb, out);
  }
}
